// GRU_Autoencoder_738734375331
// MI455X (gfx1250) — matrix operand order repaired and checked by reading, not compiled, not run on hardware
//
#include <hip/hip_runtime.h>
#include <cmath>

#define ISZ 512
#define HID 1024
#define TST 128
#define BAT 512

typedef __attribute__((ext_vector_type(16))) _Float16 v16h;
typedef __attribute__((ext_vector_type(8)))  _Float16 v8h;
typedef __attribute__((ext_vector_type(8)))  float    v8f;
typedef int v4i_vs __attribute__((vector_size(16)));   // matches builtin param

#define BLK_M 64
#define BLK_N 32
#define KU    64          /* K elements staged per pipeline round           */
#define LDAx  72          /* halves per LDS row: 64 data + 8 pad (16B mult) */
#define LDS_A (64 * LDAx) /* one A buffer (halves)                          */
#define NPIPE 3           /* async pipeline depth                           */

#if __has_builtin(__builtin_amdgcn_global_load_async_to_lds_b128)
#define HAVE_ASYNC_LDS 1
#endif

union Frag16 { v16h v; v8h h[2]; };

// ---- 16-byte global->LDS copy: async (CDNA5 ASYNCcnt path) or sync fallback -
__device__ __forceinline__ void cp16_g2l(_Float16* lds, const _Float16* g) {
#ifdef HAVE_ASYNC_LDS
  __builtin_amdgcn_global_load_async_to_lds_b128(
      (__attribute__((address_space(1))) v4i_vs*)g,
      (__attribute__((address_space(3))) v4i_vs*)lds, 0, 0);
#else
  *(v8h*)lds = *(const v8h*)g;
#endif
}
// wait until at most N async copies remain in flight (in-order retirement)
template <int N>
__device__ __forceinline__ void wait_async_n() {
#ifdef HAVE_ASYNC_LDS
#if __has_builtin(__builtin_amdgcn_s_wait_asynccnt)
  __builtin_amdgcn_s_wait_asynccnt((short)N);
#else
  asm volatile("s_wait_asynccnt %0" :: "i"(N) : "memory");
#endif
#endif
}

__device__ __forceinline__ v8h cvt8(const float4 f0, const float4 f1) {
  v8h r;
  r[0] = (_Float16)f0.x; r[1] = (_Float16)f0.y; r[2] = (_Float16)f0.z; r[3] = (_Float16)f0.w;
  r[4] = (_Float16)f1.x; r[5] = (_Float16)f1.y; r[6] = (_Float16)f1.z; r[7] = (_Float16)f1.w;
  return r;
}

// ---- staging: 64 x KU A tile ------------------------------------------------
__device__ __forceinline__ void stageA16(_Float16* As, const _Float16* A,
                                         size_t strideA, int m0, int kc, int tid) {
#pragma unroll
  for (int it = 0; it < 2; ++it) {          // 512 v8h units / 256 threads
    const int u   = tid + it * 256;
    const int row = u >> 3;
    const int seg = (u & 7) * 8;
    cp16_g2l(As + row * LDAx + seg,
             A + (size_t)(m0 + row) * strideA + (size_t)(kc + seg));
  }
}
__device__ __forceinline__ void stageA32(_Float16* As, const float* A,
                                         size_t strideA, int m0, int kc, int tid) {
#pragma unroll
  for (int it = 0; it < 2; ++it) {
    const int u   = tid + it * 256;
    const int row = u >> 3;
    const int seg = (u & 7) * 8;
    const float* p = A + (size_t)(m0 + row) * strideA + (size_t)(kc + seg);
    float4 f0 = ((const float4*)p)[0];
    float4 f1 = ((const float4*)p)[1];
    __builtin_prefetch(p + KU, 0, 1);
    *(v8h*)(As + row * LDAx + seg) = cvt8(f0, f1);
  }
}
// ---- staging: NG gate tiles of 32n x KU f16 weights -------------------------
template <int NG>
__device__ __forceinline__ void stageB16(_Float16* Bs, const _Float16* W,
                                         size_t K, int n0, int kc, int tid) {
#pragma unroll
  for (int it = 0; it < NG; ++it) {         // NG*256 v8h units
    const int u   = tid + it * 256;
    const int g   = u >> 8;
    const int rem = u & 255;
    const int row = rem >> 3;
    const int seg = (rem & 7) * 8;
    cp16_g2l(Bs + (g * 32 + row) * LDAx + seg,
             W + ((size_t)g * HID + n0 + row) * K + (size_t)(kc + seg));
  }
}

// ---- fragment loads per CDNA5 16-bit A(16x32)/B(32x16) lane layouts ---------
__device__ __forceinline__ v16h loadA(const _Float16* As, int mw, int lane, int kk) {
  const _Float16* p = As + (mw * 16 + (lane & 15)) * LDAx + kk + ((lane >> 4) * 8);
  Frag16 f;
  f.h[0] = *(const v8h*)(p);        // K = base+0..7
  f.h[1] = *(const v8h*)(p + 16);   // K = base+16..23
  return f.v;
}
__device__ __forceinline__ v16h loadB(const _Float16* Bs, int g, int nw, int lane, int kk) {
  const _Float16* p = Bs + (g * 32 + nw * 16 + (lane & 15)) * LDAx + kk + ((lane >> 4) * 8);
  Frag16 f;
  f.h[0] = *(const v8h*)(p);        // K = base+0..7
  f.h[1] = *(const v8h*)(p + 16);   // K = base+16..23
  return f.v;
}

// =============================================================================
// 3-deep pipelined multi-gate GEMM: acc[g] += A @ W_g^T
// Per round i: wait(round i copies) -> barrier -> issue round i+2 -> compute i
// =============================================================================
template <int NG, bool CONV, int KVAL>
__device__ __forceinline__ void gemm_ng(const float* A32, const _Float16* A16,
                                        size_t strideA, const _Float16* W,
                                        _Float16* AsBase, _Float16* BsBase,
                                        int m0, int n0, int tid, int mw, int nw,
                                        int lane, v8f* acc) {
  constexpr int LDS_B = NG * 32 * LDAx;
  constexpr int IPR   = NG + (CONV ? 0 : 2);   // async instrs / thread / round
  constexpr int nR    = KVAL / KU;

  auto stage = [&](int buf, int kc) {
    if (CONV) stageA32(AsBase + buf * LDS_A, A32, strideA, m0, kc, tid);
    else      stageA16(AsBase + buf * LDS_A, A16, strideA, m0, kc, tid);
    stageB16<NG>(BsBase + buf * LDS_B, W, (size_t)KVAL, n0, kc, tid);
  };

  stage(0, 0);
  if (nR > 1) stage(1, KU);

  for (int i = 0; i < nR; ++i) {
    if (i + 1 < nR) wait_async_n<IPR>();   // retire round i, keep i+1 in flight
    else            wait_async_n<0>();
    __syncthreads();
    if (i + 2 < nR) stage((i + 2) % NPIPE, (i + 2) * KU);

    const _Float16* As = AsBase + (i % NPIPE) * LDS_A;
    const _Float16* Bs = BsBase + (i % NPIPE) * LDS_B;
#pragma unroll
    for (int kk = 0; kk < KU; kk += 32) {
      v16h a = loadA(As, mw, lane, kk);
#pragma unroll
      for (int g = 0; g < NG; ++g) {
        v16h b = loadB(Bs, g, nw, lane, kk);
        acc[g] = __builtin_amdgcn_wmma_f32_16x16x32_f16(false, a, false, b,
                                                        (short)0, acc[g], false, false);
      }
    }
  }
  __syncthreads();   // protect LDS buffers before caller restages
}

// =============================================================================
// Fused GRU step; CONV=true -> A of phase 1 is fp32 (encoder x), else f16
// =============================================================================
template <bool CONV, int KX>
__global__ void __launch_bounds__(256)
gru_step_kernel(const float* __restrict__ A32, const _Float16* __restrict__ A16,
                long axStride,
                const _Float16* __restrict__ Wih, const _Float16* __restrict__ Whh,
                const float* __restrict__ bih, const float* __restrict__ bhh,
                const float* __restrict__ h32_in, const _Float16* __restrict__ h16_in,
                float* __restrict__ h32_out, _Float16* __restrict__ h16_out) {
  __shared__ _Float16 AsBuf[NPIPE * LDS_A];           // 27 KB
  __shared__ _Float16 BsBuf[NPIPE * 3 * 32 * LDAx];   // 40.5 KB

  const int tid  = threadIdx.x;
  const int lane = tid & 31;
  const int wave = tid >> 5;
  const int mw   = wave & 3;                      // 4 waves along M
  const int nw   = wave >> 2;                     // 2 waves along N
  const int n0   = blockIdx.x * BLK_N;
  const int m0   = blockIdx.y * BLK_M;

  v8f gi[3] = {};
  v8f gh[3] = {};

  gemm_ng<3, CONV, KX>(A32, A16, (size_t)axStride, Wih,
                       AsBuf, BsBuf, m0, n0, tid, mw, nw, lane, gi);
  gemm_ng<3, false, HID>(nullptr, h16_in, (size_t)HID, Whh,
                         AsBuf, BsBuf, m0, n0, tid, mw, nw, lane, gh);

  // ---- fused GRU pointwise (C/D layout: VGPR i -> row i + (lane>=16)*8) ----
  const int col   = n0 + nw * 16 + (lane & 15);
  const float br  = bih[col]       + bhh[col];
  const float bzz = bih[HID + col] + bhh[HID + col];
  const float bin = bih[2 * HID + col];
  const float bhn = bhh[2 * HID + col];
  const int rbase = m0 + mw * 16 + ((lane >> 4) * 8);
#pragma unroll
  for (int i = 0; i < 8; ++i) {
    const size_t idx = (size_t)(rbase + i) * HID + col;
    const float r  = 1.f / (1.f + __expf(-(gi[0][i] + gh[0][i] + br)));
    const float z  = 1.f / (1.f + __expf(-(gi[1][i] + gh[1][i] + bzz)));
    const float n  = tanhf(gi[2][i] + bin + r * (gh[2][i] + bhn));
    const float hn = (1.f - z) * n + z * h32_in[idx];
    h32_out[idx] = hn;
    h16_out[idx] = (_Float16)hn;
  }
}

// =============================================================================
// z = h @ Wz^T + bz  -> out[b, t, :]
// =============================================================================
__global__ void __launch_bounds__(256)
zproj_kernel(const _Float16* __restrict__ h16, const _Float16* __restrict__ Wz,
             const float* __restrict__ bz, float* __restrict__ out, long outStride) {
  __shared__ _Float16 AsBuf[NPIPE * LDS_A];
  __shared__ _Float16 BsBuf[NPIPE * 32 * LDAx];

  const int tid  = threadIdx.x;
  const int lane = tid & 31;
  const int wave = tid >> 5;
  const int mw   = wave & 3;
  const int nw   = wave >> 2;
  const int n0   = blockIdx.x * BLK_N;
  const int m0   = blockIdx.y * BLK_M;

  v8f acc[1] = {};
  gemm_ng<1, false, HID>(nullptr, h16, (size_t)HID, Wz,
                         AsBuf, BsBuf, m0, n0, tid, mw, nw, lane, acc);

  const int col   = n0 + nw * 16 + (lane & 15);
  const float bb  = bz[col];
  const int rbase = m0 + mw * 16 + ((lane >> 4) * 8);
#pragma unroll
  for (int i = 0; i < 8; ++i)
    out[(size_t)(rbase + i) * outStride + col] = acc[0][i] + bb;
}

// =============================================================================
__global__ void cvt_f16_kernel(const float* __restrict__ src,
                               _Float16* __restrict__ dst, int n) {
  const int i = (blockIdx.x * 256 + threadIdx.x) * 8;
  if (i < n) {
    float4 f0 = ((const float4*)(src + i))[0];
    float4 f1 = ((const float4*)(src + i))[1];
    *(v8h*)(dst + i) = cvt8(f0, f1);
  }
}
__global__ void fill_kernel(float* __restrict__ p32, _Float16* __restrict__ p16,
                            float v, int n) {
  const int i = blockIdx.x * 256 + threadIdx.x;
  if (i < n) { p32[i] = v; p16[i] = (_Float16)v; }
}

// =============================================================================
extern "C" void kernel_launch(void* const* d_in, const int* in_sizes, int n_in,
                              void* d_out, int out_size, void* d_ws, size_t ws_size,
                              hipStream_t stream) {
  const float* x    = (const float*)d_in[0];
  const float* eWih = (const float*)d_in[1];
  const float* eWhh = (const float*)d_in[2];
  const float* ebih = (const float*)d_in[3];
  const float* ebhh = (const float*)d_in[4];
  const float* dWih = (const float*)d_in[5];
  const float* dWhh = (const float*)d_in[6];
  const float* dbih = (const float*)d_in[7];
  const float* dbhh = (const float*)d_in[8];
  const float* Wz   = (const float*)d_in[9];
  const float* bz   = (const float*)d_in[10];
  float* out = (float*)d_out;

  // ---- workspace layout (~28 MB) ----
  char* w = (char*)d_ws;
  float*    h32A   = (float*)w;      w += (size_t)BAT * HID * 4;
  float*    h32B   = (float*)w;      w += (size_t)BAT * HID * 4;
  _Float16* h16A   = (_Float16*)w;   w += (size_t)BAT * HID * 2;
  _Float16* h16B   = (_Float16*)w;   w += (size_t)BAT * HID * 2;
  _Float16* eWih16 = (_Float16*)w;   w += (size_t)3 * HID * ISZ * 2;
  _Float16* eWhh16 = (_Float16*)w;   w += (size_t)3 * HID * HID * 2;
  _Float16* dWih16 = (_Float16*)w;   w += (size_t)3 * HID * HID * 2;
  _Float16* dWhh16 = (_Float16*)w;   w += (size_t)3 * HID * HID * 2;
  _Float16* Wz16   = (_Float16*)w;   w += (size_t)ISZ * HID * 2;

  // ---- once per launch: weights -> f16 (L2-resident afterwards) ----
  const int nWi = 3 * HID * ISZ, nWh = 3 * HID * HID, nWz = ISZ * HID;
  cvt_f16_kernel<<<(nWi / 8 + 255) / 256, 256, 0, stream>>>(eWih, eWih16, nWi);
  cvt_f16_kernel<<<(nWh / 8 + 255) / 256, 256, 0, stream>>>(eWhh, eWhh16, nWh);
  cvt_f16_kernel<<<(nWh / 8 + 255) / 256, 256, 0, stream>>>(dWih, dWih16, nWh);
  cvt_f16_kernel<<<(nWh / 8 + 255) / 256, 256, 0, stream>>>(dWhh, dWhh16, nWh);
  cvt_f16_kernel<<<(nWz / 8 + 255) / 256, 256, 0, stream>>>(Wz, Wz16, nWz);

  const int nh = BAT * HID;
  fill_kernel<<<(nh + 255) / 256, 256, 0, stream>>>(h32A, h16A, 0.1f, nh);

  const dim3 blk(256);
  const dim3 grdStep(HID / BLK_N, BAT / BLK_M);   // 32 x 8
  const dim3 grdZ(ISZ / BLK_N, BAT / BLK_M);      // 16 x 8

  float*    c32 = h32A; float*    n32 = h32B;
  _Float16* c16 = h16A; _Float16* n16 = h16B;

  // ---- encoder: h = GRU(x_t, h) ----
  for (int t = 0; t < TST; ++t) {
    gru_step_kernel<true, ISZ><<<grdStep, blk, 0, stream>>>(
        x + (size_t)t * ISZ, nullptr, (long)TST * ISZ,
        eWih16, eWhh16, ebih, ebhh, c32, c16, n32, n16);
    float* t32 = c32; c32 = n32; n32 = t32;
    _Float16* t16 = c16; c16 = n16; n16 = t16;
  }

  // ---- decoder: h = GRU(h, h); z_t = h @ Wz^T + bz ----
  for (int t = 0; t < TST; ++t) {
    gru_step_kernel<false, HID><<<grdStep, blk, 0, stream>>>(
        nullptr, c16, (long)HID,
        dWih16, dWhh16, dbih, dbhh, c32, c16, n32, n16);
    zproj_kernel<<<grdZ, blk, 0, stream>>>(
        n16, Wz16, bz, out + (size_t)t * ISZ, (long)TST * ISZ);
    float* t32 = c32; c32 = n32; n32 = t32;
    _Float16* t16 = c16; c16 = n16; n16 = t16;
  }
}
